// RotationalAttention_2302102470850
// MI455X (gfx1250) — matrix operand order repaired and checked by reading, not compiled, not run on hardware
//
#include <hip/hip_runtime.h>

// ---------------------------------------------------------------------------
// Types
// ---------------------------------------------------------------------------
typedef __bf16 v16bf __attribute__((ext_vector_type(16)));
typedef float  v8f   __attribute__((ext_vector_type(8)));
typedef float  f32x4 __attribute__((ext_vector_type(4)));
typedef int    i32x4 __attribute__((ext_vector_type(4)));
typedef unsigned int u32x2 __attribute__((ext_vector_type(2)));
typedef unsigned int u32x4 __attribute__((ext_vector_type(4)));
typedef unsigned int u32x8 __attribute__((ext_vector_type(8)));
typedef unsigned short u16;

#define B_  4
#define T_  2048
#define C_  1024
#define H_  16
#define DH_ 64
#define C3_ 3072
#define SCALE_ 0.125f

// float -> bf16 raw bits, round-to-nearest-even
static __device__ __forceinline__ u16 f2bf(float f) {
  unsigned int u = __builtin_bit_cast(unsigned int, f);
  u += 0x7fffu + ((u >> 16) & 1u);
  return (u16)(u >> 16);
}

// ---------------------------------------------------------------------------
// CDNA5 async global->LDS copy (16B per lane), ASYNCcnt-tracked.
// ---------------------------------------------------------------------------
#if __has_builtin(__builtin_amdgcn_global_load_async_to_lds_b128)
static __device__ __forceinline__ void async_cp16(const void* gsrc, void* ldst) {
  __builtin_amdgcn_global_load_async_to_lds_b128(
      (__attribute__((address_space(1))) i32x4*)gsrc,
      (__attribute__((address_space(3))) i32x4*)ldst, 0, 0);
}
#else
static __device__ __forceinline__ void async_cp16(const void* gsrc, void* ldst) {
  asm volatile("global_load_async_to_lds_b128 %0, %1, off"
               :: "v"((unsigned int)(unsigned long)ldst), "v"(gsrc)
               : "memory");
}
#endif

static __device__ __forceinline__ void wait_async0() {
#if __has_builtin(__builtin_amdgcn_s_wait_asynccnt)
  __builtin_amdgcn_s_wait_asynccnt(0);
#else
  asm volatile("s_wait_asynccnt 0x0" ::: "memory");
#endif
}

// ---------------------------------------------------------------------------
// WMMA fragment loaders from LDS (raw bf16 bits stored as u16)
// ISA 7.12.2 layouts for V_WMMA_F32_16X16X32_BF16 (wave32):
//   A 16x32: lanes 0-15 row M=lane, K{0..7}+{16..23}; lanes 16-31 K{8..15}+{24..31}
//   B 32x16: lanes 0-15 col N=lane, K{0..7}+{16..23}; lanes 16-31 col N=lane-16, K{8..15}+{24..31}
// ---------------------------------------------------------------------------
static __device__ __forceinline__ v16bf frag_a(const unsigned char* base, int row,
                                               int stride, int koff, int lane) {
  const unsigned char* p = base + row * stride + (koff + ((lane < 16) ? 0 : 8)) * 2;
  u32x4 lo = *(const u32x4*)p;
  u32x4 hi = *(const u32x4*)(p + 32);
  u32x8 c = __builtin_shufflevector(lo, hi, 0, 1, 2, 3, 4, 5, 6, 7);
  return __builtin_bit_cast(v16bf, c);
}

static __device__ __forceinline__ v16bf frag_b(const unsigned char* base, int col,
                                               int stride, int koff, int lane) {
  const unsigned char* p = base + col * stride + (koff + ((lane < 16) ? 0 : 8)) * 2;
  u32x4 lo = *(const u32x4*)p;
  u32x4 hi = *(const u32x4*)(p + 32);
  u32x8 c = __builtin_shufflevector(lo, hi, 0, 1, 2, 3, 4, 5, 6, 7);
  return __builtin_bit_cast(v16bf, c);
}

static __device__ __forceinline__ v8f wmma_bf16(v16bf a, v16bf b, v8f c) {
  return __builtin_amdgcn_wmma_f32_16x16x32_bf16(false, a, false, b, (short)0, c,
                                                 false, false);
}

// ---------------------------------------------------------------------------
// K0: f32 -> bf16 conversion, 4-wide
// ---------------------------------------------------------------------------
__global__ __launch_bounds__(256) void cvt_f32_bf16_x4(const float* __restrict__ in,
                                                       u16* __restrict__ out, int n4) {
  int i = blockIdx.x * 256 + threadIdx.x;
  if (i < n4) {
    f32x4 f = ((const f32x4*)in)[i];
    u32x2 p;
    p.x = (unsigned)f2bf(f.x) | ((unsigned)f2bf(f.y) << 16);
    p.y = (unsigned)f2bf(f.z) | ((unsigned)f2bf(f.w) << 16);
    ((u32x2*)out)[i] = p;
  }
}

// ---------------------------------------------------------------------------
// K1 / K3: out[M][N] = A[M][K] @ W[N][K]^T  (bf16 in, f32 accumulate)
// 128x128 block tile, 256 threads = 8 waves, wave tile 32x64.
// k-step 64, async double-buffered LDS staging (2 x 2 x 18KB).
// ---------------------------------------------------------------------------
#define GS 144  // GEMM LDS row stride bytes (64 bf16 + 16B pad)

template <int OUT_BF16>
__global__ __launch_bounds__(256) void gemm_xwt(const u16* __restrict__ A,
                                                const u16* __restrict__ W,
                                                void* __restrict__ outp,
                                                int N, int K,
                                                int scaleCols, float scale) {
  __shared__ __align__(16) unsigned char ldsA[2][128 * GS];
  __shared__ __align__(16) unsigned char ldsB[2][128 * GS];

  const int tid  = threadIdx.x;
  const int lane = tid & 31;
  const int w    = tid >> 5;
  const int wm   = (w & 3) * 32;
  const int wn   = (w >> 2) * 64;
  const int bm   = blockIdx.y * 128;
  const int bn   = blockIdx.x * 128;

  v8f acc[2][4] = {};

  // staging: thread -> (row r, 32-elem half), 64B of A and of B per k-step 64
  const int r = tid >> 1;
  const int h = tid & 1;
  const u16* gA = A + (long)(bm + r) * K + h * 32;
  const u16* gW = W + (long)(bn + r) * K + h * 32;
  const int sOff = r * GS + h * 64;

  // prologue: stage k-tile 0 into buffer 0
#pragma unroll
  for (int q = 0; q < 4; ++q) {
    async_cp16(gA + q * 8, ldsA[0] + sOff + q * 16);
    async_cp16(gW + q * 8, ldsB[0] + sOff + q * 16);
  }
  wait_async0();
  __syncthreads();

  for (int k0 = 0; k0 < K; k0 += 64) {
    const int cur = (k0 >> 6) & 1;
    // prefetch next k-tile into the other buffer (overlaps WMMAs)
    if (k0 + 64 < K) {
      const u16* pA = gA + k0 + 64;
      const u16* pW = gW + k0 + 64;
#pragma unroll
      for (int q = 0; q < 4; ++q) {
        async_cp16(pA + q * 8, ldsA[cur ^ 1] + sOff + q * 16);
        async_cp16(pW + q * 8, ldsB[cur ^ 1] + sOff + q * 16);
      }
    }

#pragma unroll
    for (int ks = 0; ks < 64; ks += 32) {
      v16bf af[2], bfr[4];
#pragma unroll
      for (int i = 0; i < 2; ++i)
        af[i] = frag_a(ldsA[cur], wm + i * 16 + (lane & 15), GS, ks, lane);
#pragma unroll
      for (int j = 0; j < 4; ++j)
        bfr[j] = frag_b(ldsB[cur], wn + j * 16 + (lane & 15), GS, ks, lane);
#pragma unroll
      for (int i = 0; i < 2; ++i)
#pragma unroll
        for (int j = 0; j < 4; ++j) acc[i][j] = wmma_bf16(af[i], bfr[j], acc[i][j]);
    }

    wait_async0();
    __syncthreads();
  }

  // ---- epilogue: 32-bit offset math from a single base
  const int rlo  = (lane < 16) ? 0 : 8;
  const int cn   = lane & 15;
  const int base = (bm + wm + rlo) * N + bn + wn + cn;
#pragma unroll
  for (int i = 0; i < 2; ++i)
#pragma unroll
    for (int j = 0; j < 4; ++j)
#pragma unroll
      for (int rr = 0; rr < 8; ++rr) {
        const int off = base + (i * 16 + rr) * N + j * 16;
        float v = acc[i][j][rr];
        if (OUT_BF16) {
          if (bn + wn + j * 16 + cn < scaleCols) v *= scale;
          ((u16*)outp)[off] = f2bf(v);
        } else {
          ((float*)outp)[off] = v;
        }
      }
}

// ---------------------------------------------------------------------------
// K2: fused causal attention (no softmax):
//   attn[b,h,t,:] = sum_{s<=t} (q[t].k[s]) * v[s,:]   (scale folded into q)
// grid = (T/128, B*H). 8 waves x 16 q-rows. s-chunks of 32; chunks fully
// below the diagonal take a compare-free path, diagonal chunks masked,
// above-diagonal skipped per wave.
// ---------------------------------------------------------------------------
#define QS 144   // Q LDS row stride bytes
#define KS 144   // K-chunk LDS row stride
#define VS 80    // V^T LDS row stride
#define PS 80    // P LDS row stride

__global__ __launch_bounds__(256) void attn_causal(const u16* __restrict__ qkvb,
                                                   u16* __restrict__ attnb) {
  __shared__ __align__(16) unsigned char ldsQ[128 * QS];
  __shared__ __align__(16) unsigned char ldsK[32 * KS];
  __shared__ __align__(16) unsigned char ldsVT[64 * VS];
  __shared__ __align__(16) unsigned char ldsP[8 * 16 * PS];

  const int tid  = threadIdx.x;
  const int lane = tid & 31;
  const int w    = tid >> 5;
  const int tb   = blockIdx.x;
  const int bh   = blockIdx.y;
  const int b    = bh >> 4;
  const int hh   = bh & 15;

  // ---- stage Q tile (128 x 64) once, async
  {
    const int qr = tid >> 1, half = tid & 1;
    const u16* gq = qkvb + (long)(b * T_ + tb * 128 + qr) * C3_ + hh * 64 + half * 32;
    unsigned char* dq = ldsQ + qr * QS + half * 64;
    async_cp16(gq,      dq);
    async_cp16(gq + 8,  dq + 16);
    async_cp16(gq + 16, dq + 32);
    async_cp16(gq + 24, dq + 48);
  }

  // staging assignment for K/V chunks: thread -> (s-row, 8-elem segment)
  const int srow = tid >> 3;
  const int seg  = tid & 7;
  const u16* gk = qkvb + (long)(b * T_ + srow) * C3_ + C_ + hh * 64 + seg * 8;
  const u16* gv = gk + C_;
  unsigned char* const dk = ldsK + srow * KS + seg * 16;
  u16* const vt = (u16*)ldsVT;

  wait_async0();
  __syncthreads();

  v8f acc[4] = {};
  const int nChunks  = (tb + 1) * 4;
  const int tMinWave = tb * 128 + w * 16;
  const int tMaxWave = tMinWave + 15;
  const int rlo      = (lane < 16) ? 0 : 8;
  const int cn       = lane & 15;

  for (int sc = 0; sc < nChunks; ++sc) {
    const int s0 = sc * 32;

    // ---- stage K chunk (async) + V^T chunk (VGPR transpose)
    async_cp16(gk, dk);
    {
      u32x4 vr = *(const u32x4*)gv;
      u16 tmp[8];
      __builtin_memcpy(tmp, &vr, 16);
#pragma unroll
      for (int i2 = 0; i2 < 8; ++i2)
        vt[(seg * 8 + i2) * (VS / 2) + srow] = tmp[i2];
    }
    gk += 32 * C3_;
    gv += 32 * C3_;
    wait_async0();
    __syncthreads();

    if (s0 <= tMaxWave) {  // wave-uniform causal skip
      u16* pw = (u16*)(ldsP + w * 16 * PS);
      const bool diag = (s0 + 31 > tMinWave);  // wave-uniform: chunk hits diagonal

      // ---- preload all score fragments (distinct regs -> overlapped ds loads)
      v16bf aq[2], bk[2][2];
#pragma unroll
      for (int ds = 0; ds < 2; ++ds) {
        aq[ds] = frag_a(ldsQ, w * 16 + cn, QS, ds * 32, lane);
#pragma unroll
        for (int j = 0; j < 2; ++j)
          bk[j][ds] = frag_b(ldsK, j * 16 + cn, KS, ds * 32, lane);
      }
#pragma unroll
      for (int j = 0; j < 2; ++j) {
        v8f p = wmma_bf16(aq[0], bk[j][0], v8f{});
        p = wmma_bf16(aq[1], bk[j][1], p);
        if (diag) {
          const int scol = s0 + j * 16 + cn;
#pragma unroll
          for (int rr = 0; rr < 8; ++rr) {
            const int trow = tMinWave + rr + rlo;
            float v = (scol <= trow) ? p[rr] : 0.0f;
            pw[(rr + rlo) * (PS / 2) + j * 16 + cn] = f2bf(v);
          }
        } else {
#pragma unroll
          for (int rr = 0; rr < 8; ++rr)
            pw[(rr + rlo) * (PS / 2) + j * 16 + cn] = f2bf(p[rr]);
        }
      }

      // ---- attn += P(16x32) @ v(32x64); preload all V fragments
      v16bf ap = frag_a(ldsP + w * 16 * PS, cn, PS, 0, lane);
      v16bf bv[4];
#pragma unroll
      for (int j2 = 0; j2 < 4; ++j2)
        bv[j2] = frag_b(ldsVT, j2 * 16 + cn, VS, 0, lane);
#pragma unroll
      for (int j2 = 0; j2 < 4; ++j2)
        acc[j2] = wmma_bf16(ap, bv[j2], acc[j2]);
    }
    __syncthreads();
  }

  // ---- epilogue: bf16 attn in [B*T][C], heads merged; 32-bit offsets
  const int base = (b * T_ + tb * 128 + w * 16 + rlo) * C_ + hh * 64 + cn;
#pragma unroll
  for (int j2 = 0; j2 < 4; ++j2)
#pragma unroll
    for (int rr = 0; rr < 8; ++rr)
      attnb[base + rr * C_ + j2 * 16] = f2bf(acc[j2][rr]);
}

// ---------------------------------------------------------------------------
// Host launcher
// ---------------------------------------------------------------------------
extern "C" void kernel_launch(void* const* d_in, const int* in_sizes, int n_in,
                              void* d_out, int out_size, void* d_ws, size_t ws_size,
                              hipStream_t stream) {
  (void)in_sizes; (void)n_in; (void)out_size; (void)ws_size;
  const float* x    = (const float*)d_in[0];
  const float* Wqkv = (const float*)d_in[1];
  const float* Wout = (const float*)d_in[2];
  float* out = (float*)d_out;

  unsigned char* ws = (unsigned char*)d_ws;
  size_t off = 0;
  u16* xb    = (u16*)(ws + off); off += (size_t)B_ * T_ * C_ * 2;
  u16* wqkvb = (u16*)(ws + off); off += (size_t)C3_ * C_ * 2;
  u16* woutb = (u16*)(ws + off); off += (size_t)C_ * C_ * 2;
  u16* qkvb  = (u16*)(ws + off); off += (size_t)B_ * T_ * C3_ * 2;
  u16* attnb = (u16*)(ws + off); off += (size_t)B_ * T_ * C_ * 2;

  const int nx = B_ * T_ * C_ / 4;
  const int nq = C3_ * C_ / 4;
  const int no = C_ * C_ / 4;
  cvt_f32_bf16_x4<<<dim3((nx + 255) / 256), dim3(256), 0, stream>>>(x, xb, nx);
  cvt_f32_bf16_x4<<<dim3((nq + 255) / 256), dim3(256), 0, stream>>>(Wqkv, wqkvb, nq);
  cvt_f32_bf16_x4<<<dim3((no + 255) / 256), dim3(256), 0, stream>>>(Wout, woutb, no);

  // qkv = x @ Wqkv^T (q columns scaled by 1/sqrt(Dh))
  gemm_xwt<1><<<dim3(C3_ / 128, (B_ * T_) / 128), dim3(256), 0, stream>>>(
      xb, wqkvb, qkvb, C3_, C_, /*scaleCols=*/C_, SCALE_);

  // fused causal (no-softmax) attention
  attn_causal<<<dim3(T_ / 128, B_ * H_), dim3(256), 0, stream>>>(qkvb, attnb);

  // out = attn @ Wout^T (f32 output)
  gemm_xwt<0><<<dim3(C_ / 128, (B_ * T_) / 128), dim3(256), 0, stream>>>(
      attnb, woutb, out, C_, C_, 0, 1.0f);
}
